// Attention_layer_39814346834107
// MI455X (gfx1250) — matrix operand order repaired and checked by reading, not compiled, not run on hardware
//
#include <hip/hip_runtime.h>
#include <cstdint>

#define NB    32
#define NSEQ  577
#define EMB   768
#define NH    12
#define HD    64
#define NPAD  608            // 577 padded to 19*32 for clean 32-wide K tiles
#define MTOT  (NB * NSEQ)    // 18464
#define QKVC  (3 * EMB)      // 2304
#define ATT_SCALE 0.125f     // 64^-0.5

typedef __attribute__((ext_vector_type(16))) __bf16 v16bf;
typedef __attribute__((ext_vector_type(8)))  float  v8f;

union Frag16 { uint4 u[2]; v16bf v; };

__device__ __forceinline__ v8f wmma_bf16(v16bf a, v16bf b, v8f c) {
  // D = A(16x32 bf16) * B(32x16 bf16) + C(16x16 f32)
  return __builtin_amdgcn_wmma_f32_16x16x32_bf16(false, a, false, b, (short)0, c,
                                                 false, false);
}

#if __has_builtin(__builtin_amdgcn_tensor_load_to_lds) && \
    __has_builtin(__builtin_amdgcn_s_wait_tensorcnt)
#define USE_TDM 1
typedef unsigned int u32x4 __attribute__((ext_vector_type(4)));
typedef int          i32x4 __attribute__((ext_vector_type(4)));
typedef int          i32x8 __attribute__((ext_vector_type(8)));
#endif

// ---------------------------------------------------------------------------
// Zero the N-padding rows [577,608) of Q/K and cols of V^T so the flash loop
// can iterate 32-wide K tiles unconditionally.
// ---------------------------------------------------------------------------
__global__ __launch_bounds__(256) void pad_zero(__bf16* __restrict__ Q,
                                                __bf16* __restrict__ K,
                                                __bf16* __restrict__ VT) {
  const int bh = blockIdx.x;  // B*H
  __bf16* q  = Q  + (size_t)bh * NPAD * HD;
  __bf16* k  = K  + (size_t)bh * NPAD * HD;
  __bf16* vt = VT + (size_t)bh * HD * NPAD;
  const int PADROWS = NPAD - NSEQ;  // 31
  for (int i = threadIdx.x; i < PADROWS * HD; i += blockDim.x) {
    const int r = NSEQ + i / HD;
    const int d = i % HD;
    q[(size_t)r * HD + d]    = (__bf16)0.0f;
    k[(size_t)r * HD + d]    = (__bf16)0.0f;
    vt[(size_t)d * NPAD + r] = (__bf16)0.0f;
  }
}

// ---------------------------------------------------------------------------
// QKV GEMM: Y[18464,2304] = X[18464,768] @ Wqkv[768,2304], scattered as bf16
// into Q[B,H,Npad,Dh], K[B,H,Npad,Dh], V^T[B,H,Dh,Npad].
// Block tile 128(M) x 128(N), 8 waves in 2x4, wave tile 64x32, K-step 32.
// 8 WMMAs per 12 ds_load_b128 per K-step.
// ---------------------------------------------------------------------------
__global__ __launch_bounds__(256) void qkv_gemm(const float* __restrict__ X,
                                                const float* __restrict__ W,
                                                __bf16* __restrict__ Qo,
                                                __bf16* __restrict__ Ko,
                                                __bf16* __restrict__ VTo) {
  __shared__ __bf16 As[128 * 40];  // [128 rows M][32 K] stride 40 (16B-aligned)
  __shared__ __bf16 Bs[128 * 40];  // transposed: [128 cols N][32 K] stride 40

  const int t = threadIdx.x;
  const int wave = t >> 5, lane = t & 31, l16 = lane & 15, lh = lane >> 4;
  const int wm = wave >> 2, wn = wave & 3;  // 2 x 4 wave grid
  const int m0 = blockIdx.x * 128, n0 = blockIdx.y * 128;

  v8f acc[4][2] = {};

  const int ar = t >> 3, ac = (t & 7) * 4;   // A staging: rows ar+{0,32,64,96}
  const int bk = t >> 5, bc = (t & 31) * 4;  // B staging

  for (int k0 = 0; k0 < EMB; k0 += 32) {
    __syncthreads();
    // stage A (fp32 -> bf16)
#pragma unroll
    for (int rr = 0; rr < 4; rr++) {
      int row = m0 + ar + rr * 32;
      if (row >= MTOT) row = MTOT - 1;
      const float4 xv = *(const float4*)(X + (size_t)row * EMB + k0 + ac);
      __bf16* dst = &As[(ar + rr * 32) * 40 + ac];
      dst[0] = (__bf16)xv.x; dst[1] = (__bf16)xv.y;
      dst[2] = (__bf16)xv.z; dst[3] = (__bf16)xv.w;
    }
    // stage B transposed (fp32 -> bf16)
#pragma unroll
    for (int i = 0; i < 4; i++) {
      const int k = bk + i * 8;
      const float4 wv = *(const float4*)(W + (size_t)(k0 + k) * QKVC + n0 + bc);
      Bs[(bc + 0) * 40 + k] = (__bf16)wv.x;
      Bs[(bc + 1) * 40 + k] = (__bf16)wv.y;
      Bs[(bc + 2) * 40 + k] = (__bf16)wv.z;
      Bs[(bc + 3) * 40 + k] = (__bf16)wv.w;
    }
    if (k0 + 32 < EMB) {  // gfx1250 global_prefetch_b8 for next tiles
      __builtin_prefetch(X + (size_t)(m0 + ar) * EMB + k0 + 32 + ac, 0, 1);
      __builtin_prefetch(W + (size_t)(k0 + 32 + bk) * QKVC + n0 + bc, 0, 1);
    }
    __syncthreads();

    v16bf af[4], bfr[2];
#pragma unroll
    for (int mi = 0; mi < 4; mi++) {  // A frag: row l16, chunks {8h..},{16+8h..}
      const int row = wm * 64 + mi * 16 + l16;
      Frag16 f;
      f.u[0] = *(const uint4*)&As[row * 40 + 8 * lh];
      f.u[1] = *(const uint4*)&As[row * 40 + 16 + 8 * lh];
      af[mi] = f.v;
    }
#pragma unroll
    for (int ni = 0; ni < 2; ni++) {  // B frag: col l16, chunks {8h..},{16+8h..}
      const int col = wn * 32 + ni * 16 + l16;
      Frag16 f;
      f.u[0] = *(const uint4*)&Bs[col * 40 + 8 * lh];
      f.u[1] = *(const uint4*)&Bs[col * 40 + 16 + 8 * lh];
      bfr[ni] = f.v;
    }
#pragma unroll
    for (int mi = 0; mi < 4; mi++)
#pragma unroll
      for (int ni = 0; ni < 2; ni++)
        acc[mi][ni] = wmma_bf16(af[mi], bfr[ni], acc[mi][ni]);
  }

  // epilogue: scatter to Q / K / V^T (bf16)
#pragma unroll
  for (int mi = 0; mi < 4; mi++) {
#pragma unroll
    for (int ni = 0; ni < 2; ni++) {
      const int col = n0 + wn * 32 + ni * 16 + l16;
      const int s = col / EMB;  // 0=Q 1=K 2=V
      const int rem = col % EMB;
      const int hh = rem >> 6, d = rem & 63;
#pragma unroll
      for (int j = 0; j < 8; j++) {  // C layout: row = j + 8*(lane/16)
        const int row = m0 + wm * 64 + mi * 16 + j + 8 * lh;
        if (row < MTOT) {
          const int b = row / NSEQ, n = row % NSEQ;
          const size_t bh = (size_t)b * NH + hh;
          const __bf16 val = (__bf16)acc[mi][ni][j];
          if (s == 0)      Qo[(bh * NPAD + n) * HD + d]  = val;
          else if (s == 1) Ko[(bh * NPAD + n) * HD + d]  = val;
          else             VTo[(bh * HD + d) * NPAD + n] = val;
        }
      }
    }
  }
}

// ---------------------------------------------------------------------------
// Flash attention: one wave per 16-row Q tile, online softmax, WMMA for both
// QK^T and PV. P transits per-wave LDS to convert C-layout -> A-fragment.
// Grid: (B*H, ceil(37/8)=5), block = 256 (8 waves).
// ---------------------------------------------------------------------------
__global__ __launch_bounds__(256) void flash_attn(const __bf16* __restrict__ Q,
                                                  const __bf16* __restrict__ K,
                                                  const __bf16* __restrict__ VT,
                                                  __bf16* __restrict__ Attn) {
  __shared__ __bf16 Pshared[8 * 16 * 40];  // per-wave 16x32 P tile, stride 40

  const int t = threadIdx.x;
  const int wave = t >> 5, lane = t & 31, l16 = lane & 15, lh = lane >> 4;
  const int bh = blockIdx.x;
  const int b = bh / NH, h = bh % NH;
  const int qtile = blockIdx.y * 8 + wave;
  const int NQT = (NSEQ + 15) / 16;  // 37
  if (qtile >= NQT) return;          // wave-uniform exit, no barriers used
  const int qb = qtile * 16;

  const __bf16* Qb = Q + (size_t)bh * NPAD * HD;
  const __bf16* Kb = K + (size_t)bh * NPAD * HD;
  const __bf16* Vb = VT + (size_t)bh * HD * NPAD;
  __bf16* Pl = Pshared + wave * 16 * 40;

  // Q A-frags for dh chunks [0,32) and [32,64): row l16, contiguous 16h..16h+15
  v16bf qf[2];
#pragma unroll
  for (int c = 0; c < 2; c++) {
    const __bf16* p = Qb + (size_t)(qb + l16) * HD + c * 32 + 16 * lh;
    Frag16 f;
    f.u[0] = *(const uint4*)p;
    f.u[1] = *(const uint4*)(p + 8);
    qf[c] = f.v;
  }

  v8f o[4] = {};
  float mr[8], lr[8];
#pragma unroll
  for (int j = 0; j < 8; j++) { mr[j] = -1e30f; lr[j] = 0.0f; }

  for (int kb = 0; kb < NPAD; kb += 32) {
    // ---- S = Q @ K^T (scaled), two 16x16 tiles covering k cols kb..kb+31 ----
    v8f s0 = {}, s1 = {};
#pragma unroll
    for (int tt = 0; tt < 2; tt++) {
      // B frag from K[row=kb+tt*16+l16][dh = c*32 + 16h + 0..15], contiguous
      const __bf16* kp = Kb + (size_t)(kb + tt * 16 + l16) * HD + 16 * lh;
      Frag16 f0, f1;
      f0.u[0] = *(const uint4*)kp;        f0.u[1] = *(const uint4*)(kp + 8);
      f1.u[0] = *(const uint4*)(kp + 32); f1.u[1] = *(const uint4*)(kp + 40);
      if (tt == 0) { s0 = wmma_bf16(qf[0], f0.v, s0); s0 = wmma_bf16(qf[1], f1.v, s0); }
      else         { s1 = wmma_bf16(qf[0], f0.v, s1); s1 = wmma_bf16(qf[1], f1.v, s1); }
    }
    const int col0 = kb + l16;
#pragma unroll
    for (int j = 0; j < 8; j++) { s0[j] *= ATT_SCALE; s1[j] *= ATT_SCALE; }
    if (col0 >= NSEQ) {
#pragma unroll
      for (int j = 0; j < 8; j++) s0[j] = -1e30f;
    }
    if (col0 + 16 >= NSEQ) {
#pragma unroll
      for (int j = 0; j < 8; j++) s1[j] = -1e30f;
    }

    // ---- online softmax per row (rows j+8h live across 16-lane groups) ----
#pragma unroll
    for (int j = 0; j < 8; j++) {
      float v = fmaxf(s0[j], s1[j]);
      v = fmaxf(v, __shfl_xor(v, 1));
      v = fmaxf(v, __shfl_xor(v, 2));
      v = fmaxf(v, __shfl_xor(v, 4));
      v = fmaxf(v, __shfl_xor(v, 8));
      const float mn = fmaxf(mr[j], v);
      const float al = __expf(mr[j] - mn);
      mr[j] = mn;
      const float p0 = __expf(s0[j] - mn);
      const float p1 = __expf(s1[j] - mn);
      s0[j] = p0; s1[j] = p1;
      float rs = p0 + p1;
      rs += __shfl_xor(rs, 1);
      rs += __shfl_xor(rs, 2);
      rs += __shfl_xor(rs, 4);
      rs += __shfl_xor(rs, 8);
      lr[j] = lr[j] * al + rs;
      o[0][j] *= al; o[1][j] *= al; o[2][j] *= al; o[3][j] *= al;
    }

    // ---- P (C-layout) -> LDS -> A-fragment layout ----
#pragma unroll
    for (int j = 0; j < 8; j++) {
      Pl[(j + 8 * lh) * 40 + l16]      = (__bf16)s0[j];
      Pl[(j + 8 * lh) * 40 + 16 + l16] = (__bf16)s1[j];
    }
    asm volatile("s_wait_dscnt 0x0" ::: "memory");  // intra-wave LDS RAW
    Frag16 pf;
    const __bf16* pp = Pl + l16 * 40 + 16 * lh;
    pf.u[0] = *(const uint4*)pp;
    pf.u[1] = *(const uint4*)(pp + 8);

    // ---- O += P @ V  (V^T layout -> contiguous B-frag loads) ----
#pragma unroll
    for (int t2 = 0; t2 < 4; t2++) {
      const __bf16* vp = Vb + (size_t)(t2 * 16 + l16) * NPAD + kb + 16 * lh;
      Frag16 vf;
      vf.u[0] = *(const uint4*)vp;
      vf.u[1] = *(const uint4*)(vp + 8);
      o[t2] = wmma_bf16(pf.v, vf.v, o[t2]);
    }
  }

  // ---- finalize: divide by row sums, store bf16 to attn[B,N,E] ----
#pragma unroll
  for (int j = 0; j < 8; j++) {
    const float inv = lr[j] > 0.0f ? 1.0f / lr[j] : 0.0f;
    o[0][j] *= inv; o[1][j] *= inv; o[2][j] *= inv; o[3][j] *= inv;
  }
#pragma unroll
  for (int t2 = 0; t2 < 4; t2++) {
#pragma unroll
    for (int j = 0; j < 8; j++) {
      const int q = qb + j + 8 * lh;
      if (q < NSEQ)
        Attn[((size_t)b * NSEQ + q) * EMB + h * HD + t2 * 16 + l16] =
            (__bf16)o[t2][j];
    }
  }
}

// ---------------------------------------------------------------------------
// Projection GEMM: Out[18464,768] = Attn_bf16 @ Wproj[768,768] + bias (fp32).
// A tile (128x32 bf16, row-major) staged via the Tensor Data Mover: LDS pad
// (interval=16 DW, amount=4 DW) reproduces the 40-element padded row stride;
// OOB rows beyond tensor_dim1 read as zero (epilogue masks those rows anyway).
// ---------------------------------------------------------------------------
__global__ __launch_bounds__(256) void proj_gemm(const __bf16* __restrict__ A,
                                                 const float* __restrict__ W,
                                                 const float* __restrict__ Bias,
                                                 float* __restrict__ Out) {
  __shared__ __bf16 As[128 * 40];
  __shared__ __bf16 Bs[128 * 40];

  const int t = threadIdx.x;
  const int wave = t >> 5, lane = t & 31, l16 = lane & 15, lh = lane >> 4;
  const int wm = wave >> 2, wn = wave & 3;
  const int m0 = blockIdx.x * 128, n0 = blockIdx.y * 128;

  v8f acc[4][2] = {};

  const int bk = t >> 5, bc = (t & 31) * 4;  // B staging

  for (int k0 = 0; k0 < EMB; k0 += 32) {
    __syncthreads();  // all waves done reading LDS before overwrite
#ifdef USE_TDM
    if (wave == 0) {
      // D# group 0: count=1 | lds_addr | global_addr | type=2
      const uint64_t gaddr =
          (uint64_t)(uintptr_t)A + ((uint64_t)m0 * EMB + k0) * 2u;
      const uint32_t lds = (uint32_t)(uintptr_t)(void*)As;  // LDS byte offset
      u32x4 g0;
      g0[0] = 1u;                                       // count = 1 (valid)
      g0[1] = lds;                                      // lds_addr
      g0[2] = (uint32_t)gaddr;                          // global_addr[31:0]
      g0[3] = (uint32_t)(gaddr >> 32) | 0x80000000u;    // ga[56:32] | type=2
      // D# group 1: data_size=2B, pad_enable, pad_interval=16DW, pad_amount=4DW
      i32x8 g1;
      g1[0] = (int)((1u << 16) | (1u << 20) | (3u << 22) | (3u << 25));
      g1[1] = (int)(768u << 16);       // tensor_dim0 = 768 (bits 79:48)
      g1[2] = (int)((uint32_t)MTOT << 16);  // tensor_dim1 = 18464 (bits 111:80)
      g1[3] = (int)(32u << 16);        // tile_dim0 = 32 (bits 127:112)
      g1[4] = 128;                     // tile_dim1 = 128 (bits 143:128)
      g1[5] = 768;                     // tensor_dim0_stride = 768 (bits 207:160)
      g1[6] = 0;
      g1[7] = 0;
      const i32x4 z4 = {0, 0, 0, 0};   // 2D tensor: groups 2/3 unused
      const i32x8 z8 = {0, 0, 0, 0, 0, 0, 0, 0};
      __builtin_amdgcn_tensor_load_to_lds(g0, g1, z4, z4, z8, 0);
    }
#else
    {  // manual A staging: 128 rows x 32 bf16, 16 elems per thread
      const int ar2 = t >> 1, ac2 = (t & 1) * 16;
      int row = m0 + ar2;
      if (row >= MTOT) row = MTOT - 1;
      const uint4 a0 = *(const uint4*)(A + (size_t)row * EMB + k0 + ac2);
      const uint4 a1 = *(const uint4*)(A + (size_t)row * EMB + k0 + ac2 + 8);
      const __bf16* a8 = (const __bf16*)&a0;
      const __bf16* a9 = (const __bf16*)&a1;
      __bf16* dst = &As[ar2 * 40 + ac2];
#pragma unroll
      for (int x = 0; x < 8; x++) { dst[x] = a8[x]; dst[x + 8] = a9[x]; }
    }
#endif
    // stage B transposed (fp32 -> bf16)
#pragma unroll
    for (int i = 0; i < 4; i++) {
      const int k = bk + i * 8;
      const float4 wv = *(const float4*)(W + (size_t)(k0 + k) * EMB + n0 + bc);
      Bs[(bc + 0) * 40 + k] = (__bf16)wv.x;
      Bs[(bc + 1) * 40 + k] = (__bf16)wv.y;
      Bs[(bc + 2) * 40 + k] = (__bf16)wv.z;
      Bs[(bc + 3) * 40 + k] = (__bf16)wv.w;
    }
    if (k0 + 32 < EMB)
      __builtin_prefetch(W + (size_t)(k0 + 32 + bk) * EMB + n0 + bc, 0, 1);
#ifdef USE_TDM
    if (wave == 0) __builtin_amdgcn_s_wait_tensorcnt((short)0);
#endif
    __syncthreads();  // TDM data + B stores visible to all waves

    v16bf af[4], bfr[2];
#pragma unroll
    for (int mi = 0; mi < 4; mi++) {
      const int row = wm * 64 + mi * 16 + l16;
      Frag16 f;
      f.u[0] = *(const uint4*)&As[row * 40 + 8 * lh];
      f.u[1] = *(const uint4*)&As[row * 40 + 16 + 8 * lh];
      af[mi] = f.v;
    }
#pragma unroll
    for (int ni = 0; ni < 2; ni++) {
      const int col = wn * 32 + ni * 16 + l16;
      Frag16 f;
      f.u[0] = *(const uint4*)&Bs[col * 40 + 8 * lh];
      f.u[1] = *(const uint4*)&Bs[col * 40 + 16 + 8 * lh];
      bfr[ni] = f.v;
    }
#pragma unroll
    for (int mi = 0; mi < 4; mi++)
#pragma unroll
      for (int ni = 0; ni < 2; ni++)
        acc[mi][ni] = wmma_bf16(af[mi], bfr[ni], acc[mi][ni]);
  }

#pragma unroll
  for (int mi = 0; mi < 4; mi++) {
#pragma unroll
    for (int ni = 0; ni < 2; ni++) {
      const int col = n0 + wn * 32 + ni * 16 + l16;
      const float bias = Bias[col];
#pragma unroll
      for (int j = 0; j < 8; j++) {
        const int row = m0 + wm * 64 + mi * 16 + j + 8 * lh;
        if (row < MTOT)
          Out[(size_t)row * EMB + col] = acc[mi][ni][j] + bias;
      }
    }
  }
}

// ---------------------------------------------------------------------------
extern "C" void kernel_launch(void* const* d_in, const int* in_sizes, int n_in,
                              void* d_out, int out_size, void* d_ws, size_t ws_size,
                              hipStream_t stream) {
  (void)in_sizes; (void)n_in; (void)out_size; (void)ws_size;
  const float* x      = (const float*)d_in[0];  // [32,577,768]
  const float* w_qkv  = (const float*)d_in[1];  // [768,2304]
  const float* w_proj = (const float*)d_in[2];  // [768,768]
  const float* b_proj = (const float*)d_in[3];  // [768]
  float* out = (float*)d_out;                   // [32,577,768] fp32

  // workspace layout (bf16): Q, K, V^T padded to NPAD, attn output
  const size_t per = (size_t)NB * NH * NPAD * HD;  // 14,942,208 elems
  __bf16* qw    = (__bf16*)d_ws;
  __bf16* kw    = qw + per;
  __bf16* vtw   = kw + per;
  __bf16* attnw = vtw + per;                       // MTOT*EMB elems

  pad_zero<<<NB * NH, 256, 0, stream>>>(qw, kw, vtw);
  qkv_gemm<<<dim3((MTOT + 127) / 128, QKVC / 128), 256, 0, stream>>>(
      x, w_qkv, qw, kw, vtw);
  flash_attn<<<dim3(NB * NH, 5), 256, 0, stream>>>(qw, kw, vtw, attnw);
  proj_gemm<<<dim3((MTOT + 127) / 128, EMB / 128), 256, 0, stream>>>(
      attnw, w_proj, b_proj, out);
}
